// FeedForward_76871324663965
// MI455X (gfx1250) — matrix operand order repaired and checked by reading, not compiled, not run on hardware
//
#include <hip/hip_runtime.h>

// Problem constants (match reference)
#define BB 2
#define SS 1024
#define DD 1024
#define HH 2048
#define EE 8
#define KK 2

constexpr int NTOK     = BB * SS;              // 2048 tokens
constexpr int NASSIGN  = NTOK * KK;            // 4096 (token, slot) assignments
constexpr int CAP      = NASSIGN;              // worst-case per-expert list capacity
constexpr int MAX_TILES = (NASSIGN + 15) / 16; // 256 tiles of 16 tokens
constexpr int HC  = 512;                       // H-chunk held in LDS per iteration
constexpr int LDX = DD + 8;                    // padded LDS strides (bank spread)
constexpr int LDH = HC + 8;

typedef __attribute__((ext_vector_type(16))) __bf16 v16bf;
typedef __attribute__((ext_vector_type(8)))  __bf16 v8bf;
typedef __attribute__((ext_vector_type(4)))  __bf16 v4bf;
typedef __attribute__((ext_vector_type(8)))  float  v8f;

static __device__ __forceinline__ v16bf mk16(v8bf lo, v8bf hi) {
    return __builtin_shufflevector(lo, hi, 0,1,2,3,4,5,6,7,8,9,10,11,12,13,14,15);
}

// jax.nn.gelu approximate=True, branch-free via hardware transcendentals:
//   0.5*x*(1+tanh(u)) == x * sigmoid(2u),  u = 0.7978845608*(x + 0.044715*x^3)
//   sigmoid(t) = 1/(1 + exp2(-log2(e)*t))
// Tails are exact: exp2 -> 0 => g = x;  exp2 -> +inf => g = 0. No NaN, no branches.
static __device__ __forceinline__ float gelu_tanh(float x) {
    float x3 = x * x * x;
    float t  = 1.595769122f * x + 0.0713548162726f * x3;   // 2u
    float e  = __builtin_amdgcn_exp2f(-1.442695041f * t);  // exp(-2u)
    float s  = __builtin_amdgcn_rcpf(1.0f + e);            // sigmoid(2u)
    return x * s;
}

// ---------------- helper kernels ----------------

__global__ void zero_f32(float* p, int n) {
    int i = blockIdx.x * blockDim.x + threadIdx.x;
    if (i < n) p[i] = 0.0f;
}

__global__ void zero_i32(int* p, int n) {
    int i = blockIdx.x * blockDim.x + threadIdx.x;
    if (i < n) p[i] = 0;
}

// f32 -> bf16, 4 elements/thread, vector load + packed 64-bit store. n % 4 == 0.
__global__ void cvt_bf16(const float* __restrict__ s, __bf16* __restrict__ d, int n) {
    int i = 4 * (blockIdx.x * blockDim.x + threadIdx.x);
    int stride = 4 * gridDim.x * blockDim.x;
    for (; i < n; i += stride) {
        float4 v = *(const float4*)(s + i);
        v4bf o = { (__bf16)v.x, (__bf16)v.y, (__bf16)v.z, (__bf16)v.w };
        *(v4bf*)(d + i) = o;
    }
}

__global__ void scatter_kernel(const int* __restrict__ eidx, int* __restrict__ cnt,
                               int* __restrict__ lists, int n) {
    int i = blockIdx.x * blockDim.x + threadIdx.x;
    if (i < n) {
        int e = eidx[i];
        int p = atomicAdd(&cnt[e], 1);
        lists[e * CAP + p] = i;   // assignment index: token = i/KK, weight at ew[i]
    }
}

// ---------------- main grouped-GEMM MoE kernel ----------------
// Block = 16 tokens of one expert. 256 threads = 8 waves (wave32).
__global__ __launch_bounds__(256) void moe_ffn_kernel(
    const float*  __restrict__ x,      // [NTOK, DD] f32
    const __bf16* __restrict__ kb,     // [EE, HH, DD] bf16
    const __bf16* __restrict__ vb,     // [EE, DD, HH] bf16
    const float*  __restrict__ ew,     // [NASSIGN] f32 weights
    const int*    __restrict__ lists,  // [EE, CAP]
    const int*    __restrict__ cnt,    // [EE]
    float*        __restrict__ out)    // [NTOK, DD] f32 (pre-zeroed)
{
    const int e    = blockIdx.x >> 8;        // / MAX_TILES
    const int tile = blockIdx.x & (MAX_TILES - 1);
    const int n_e  = cnt[e];
    if (tile * 16 >= n_e) return;

    __shared__ __align__(32) __bf16 xs[16 * LDX];  // x tile, bf16
    __shared__ __align__(32) __bf16 hs[16 * LDH];  // gelu(hidden) chunk, bf16
    __shared__ int   toks[16];
    __shared__ float wts[16];

    const int tid = threadIdx.x;

    // Per-row metadata: token id + combine weight (0 for padded rows).
    if (tid < 16) {
        int idx = tile * 16 + tid;
        if (idx < n_e) {
            int a = lists[e * CAP + idx];
            toks[tid] = a >> 1;       // KK == 2
            wts[tid]  = ew[a];
        } else {
            toks[tid] = 0;
            wts[tid]  = 0.0f;
        }
    }
    __syncthreads();

    // Gather 16 x-rows (f32 -> bf16) into LDS. 16 threads per row, 64 floats each.
    {
        int r = tid >> 4, l = tid & 15;
        const float* src = x + (size_t)toks[r] * DD + l * 64;
        __bf16* dst = xs + r * LDX + l * 64;
        #pragma unroll
        for (int j = 0; j < 64; j += 4) {
            float4 v = *(const float4*)(src + j);
            v4bf o = { (__bf16)v.x, (__bf16)v.y, (__bf16)v.z, (__bf16)v.w };
            *(v4bf*)(dst + j) = o;
        }
    }
    __syncthreads();

    const int wave = tid >> 5;
    const int lane = tid & 31;
    const int ln   = lane & 15;                 // N column / A row within tile
    const int kbA  = (lane < 16) ? 0 : 16;      // A-frag k-half (contiguous k, matches B)
    const int kbB  = (lane < 16) ? 0 : 16;      // B-frag k-half (contiguous k, matches A)
    const int mb   = (lane < 16) ? 0 : 8;       // C/D row base

    const __bf16* kbE = kb + (size_t)e * HH * DD;
    const __bf16* vbE = vb + (size_t)e * DD * HH;

    v8f acc[8];
    const v8f vzero = {0.f,0.f,0.f,0.f,0.f,0.f,0.f,0.f};
    #pragma unroll
    for (int j = 0; j < 8; ++j) acc[j] = vzero;

    for (int hcb = 0; hcb < HH; hcb += HC) {
        // ---- GEMM1: hidden[16 x HC] = x[16 x DD] * keys^T, gelu -> LDS ----
        for (int th = wave; th < HC / 16; th += 8) {
            const int hg = hcb + th * 16;
            v8f c = vzero;
            const __bf16* arow = xs + ln * LDX + kbA;
            const __bf16* brow = kbE + (size_t)(hg + ln) * DD + kbB;
            #pragma unroll 4
            for (int d0 = 0; d0 < DD; d0 += 32) {
                v16bf a = mk16(*(const v8bf*)(arow + d0),
                               *(const v8bf*)(arow + d0 + 8));
                v16bf b = *(const v16bf*)(brow + d0);
                c = __builtin_amdgcn_wmma_f32_16x16x32_bf16(
                        false, a, false, b, (short)0, c, false, false);
            }
            #pragma unroll
            for (int r = 0; r < 8; ++r)
                hs[(mb + r) * LDH + th * 16 + ln] = (__bf16)gelu_tanh(c[r]);
        }
        __syncthreads();

        // ---- GEMM2: out[16 x DD] += hidden[16 x HC] * values^T (accumulate) ----
        #pragma unroll
        for (int j = 0; j < 8; ++j) {
            const int d0 = (wave * 8 + j) * 16;
            const __bf16* arow = hs + ln * LDH + kbA;
            const __bf16* brow = vbE + (size_t)(d0 + ln) * HH + hcb + kbB;
            v8f c = acc[j];
            #pragma unroll 4
            for (int h0 = 0; h0 < HC; h0 += 32) {
                v16bf a = mk16(*(const v8bf*)(arow + h0),
                               *(const v8bf*)(arow + h0 + 8));
                v16bf b = *(const v16bf*)(brow + h0);
                c = __builtin_amdgcn_wmma_f32_16x16x32_bf16(
                        false, a, false, b, (short)0, c, false, false);
            }
            acc[j] = c;
        }
        __syncthreads();
    }

    // ---- weighted combine into global out (token may appear under 2 experts) ----
    #pragma unroll
    for (int j = 0; j < 8; ++j) {
        const int d0 = (wave * 8 + j) * 16;
        #pragma unroll
        for (int r = 0; r < 8; ++r) {
            const int m = mb + r;
            unsafeAtomicAdd(out + (size_t)toks[m] * DD + d0 + ln, wts[m] * acc[j][r]);
        }
    }
}

// ---------------- host launcher ----------------

extern "C" void kernel_launch(void* const* d_in, const int* in_sizes, int n_in,
                              void* d_out, int out_size, void* d_ws, size_t ws_size,
                              hipStream_t stream) {
    const float* x      = (const float*)d_in[0];   // [B,S,D]
    const float* keys   = (const float*)d_in[1];   // [E,H,D]
    const float* values = (const float*)d_in[2];   // [E,D,H]
    const int*   eidx   = (const int*)  d_in[3];   // [B,S,K]
    const float* ew     = (const float*)d_in[4];   // [B,S,K]
    float* out = (float*)d_out;

    // Workspace layout: bf16 keys | bf16 values | counts | per-expert lists
    const size_t wcnt = (size_t)EE * HH * DD;      // 16,777,216 elements each
    __bf16* kb    = (__bf16*)d_ws;
    __bf16* vb    = kb + wcnt;
    int*    cnt   = (int*)(vb + wcnt);
    int*    lists = cnt + 16;

    zero_f32<<<(NTOK * DD + 255) / 256, 256, 0, stream>>>(out, NTOK * DD);
    zero_i32<<<1, 64, 0, stream>>>(cnt, EE);
    cvt_bf16<<<2048, 256, 0, stream>>>(keys,   kb, (int)wcnt);
    cvt_bf16<<<2048, 256, 0, stream>>>(values, vb, (int)wcnt);
    scatter_kernel<<<(NASSIGN + 255) / 256, 256, 0, stream>>>(eidx, cnt, lists, NASSIGN);
    moe_ffn_kernel<<<EE * MAX_TILES, 256, 0, stream>>>(x, kb, vb, ew, lists, cnt, out);
}
